// SeqClassLSTM_44727789420875
// MI455X (gfx1250) — matrix operand order repaired and checked by reading, not compiled, not run on hardware
//
#include <hip/hip_runtime.h>

#define TLEN   8192
#define HID    32
#define CT     64            // time-steps staged per LDS chunk
#define ROWW   (CT + 4)      // LDS row stride in dwords (padded: rows hit distinct banks)
#define NCHUNK (TLEN / CT)

typedef __attribute__((ext_vector_type(16))) _Float16 v16h;
typedef __attribute__((ext_vector_type(8)))  float    v8f;
typedef __attribute__((ext_vector_type(2)))  _Float16 h2;
typedef __attribute__((ext_vector_type(8)))  unsigned v8u;
typedef __attribute__((ext_vector_type(4)))  unsigned v4u;
typedef __attribute__((ext_vector_type(8)))  int      v8i;
typedef __attribute__((ext_vector_type(4)))  int      v4i;

#if __has_builtin(__builtin_amdgcn_tensor_load_to_lds) && __has_builtin(__builtin_amdgcn_s_wait_tensorcnt)
#define USE_TDM 1
#else
#define USE_TDM 0
#endif

// Split workgroup barrier: signal early, wait late (hide latency behind c-init math).
#if __has_builtin(__builtin_amdgcn_s_barrier_signal) && __has_builtin(__builtin_amdgcn_s_barrier_wait)
#define BAR_SIGNAL()                                         \
  do {                                                       \
    __builtin_amdgcn_fence(__ATOMIC_RELEASE, "workgroup");   \
    __builtin_amdgcn_s_barrier_signal(-1);                   \
  } while (0)
#define BAR_WAIT()                                           \
  do {                                                       \
    __builtin_amdgcn_s_barrier_wait(-1);                     \
    __builtin_amdgcn_fence(__ATOMIC_ACQUIRE, "workgroup");   \
  } while (0)
#else
#define BAR_SIGNAL() do { } while (0)
#define BAR_WAIT()   __syncthreads()
#endif

__device__ __forceinline__ float fast_tanh(float x) {
#if __has_builtin(__builtin_amdgcn_tanhf)
  return __builtin_amdgcn_tanhf(x);
#else
  float a = __builtin_fminf(-2.885390082f * x, 80.0f);
  float t = __builtin_amdgcn_exp2f(a);
  return (1.0f - t) * __builtin_amdgcn_rcpf(1.0f + t);
#endif
}
__device__ __forceinline__ float fast_sigm(float x) {
  return __builtin_fmaf(0.5f, fast_tanh(0.5f * x), 0.5f);
}

#if USE_TDM
// 2D TDM descriptor: tile = CT x 16 f32, row stride = TLEN elements,
// LDS pad: 4 dwords after every 64 dwords (pad_interval code 5, pad_amount code 3).
__device__ __forceinline__ void tdm_load_tile(const float* gsrc, unsigned lds_byte_off) {
  unsigned long long ga = (unsigned long long)gsrc;
  v4u g0;
  g0[0] = 1u;                                                   // count=1 (valid descriptor)
  g0[1] = lds_byte_off;                                         // lds_addr
  g0[2] = (unsigned)ga;                                         // global_addr[31:0]
  g0[3] = ((unsigned)(ga >> 32) & 0x01FFFFFFu) | (2u << 30);    // global_addr[56:32] | type=2
  v8i g1;
  g1[0] = (2 << 16) | (1 << 20) | (5 << 22) | (3 << 25);        // data_size=4B, pad_en, int=64dw, amt=4dw
  g1[1] = (int)((TLEN & 0xFFFF) << 16);                         // tensor_dim0[15:0]
  g1[2] = (int)(16 << 16);                                      // tensor_dim1 = 16 rows
  g1[3] = (int)(CT << 16);                                      // tile_dim0 = CT
  g1[4] = 16;                                                   // tile_dim1 = 16
  g1[5] = TLEN;                                                 // tensor_dim0_stride (elements)
  g1[6] = 0;
  g1[7] = 0;
  v4i z4 = {0, 0, 0, 0};
  v8i z8 = {0, 0, 0, 0, 0, 0, 0, 0};
  __builtin_amdgcn_tensor_load_to_lds(g0, g1, z4, z4, z8, 0);
}
#else
__device__ __forceinline__ void stage_chunk(float* dst, const float* xrow, int t0, int tid) {
  // 16 rows x CT floats = 256 float4; 4 per thread (64 threads), coalesced per row.
#pragma unroll
  for (int i = 0; i < 4; ++i) {
    int idx = tid + i * 64;
    int row = idx >> 4;
    int t4  = idx & 15;
    float4 v = *(const float4*)(xrow + (unsigned long long)row * TLEN + t0 + 4 * t4);
    *(float4*)(dst + row * ROWW + 4 * t4) = v;
  }
}
#endif

// 2 waves per block; wave w owns hidden chunk [16w, 16w+16) = one gate tile per i/f/g/o.
__global__ __launch_bounds__(64) void lstm_wmma_split_kernel(
    const float* __restrict__ x, const float* __restrict__ W_ih,
    const float* __restrict__ W_hh, const float* __restrict__ b_ih,
    const float* __restrict__ b_hh, const float* __restrict__ fc_w,
    const float* __restrict__ fc_b, float* __restrict__ out) {
  __shared__ float xls[2][16 * ROWW];                       // x staging, double buffered
  __shared__ __align__(16) unsigned xch[2][4][16][4];       // h^T exchange [parity][2w+half][m][j]
  __shared__ float fpart[2][64];                            // final FC partials

  const int tid    = threadIdx.x;
  const int wib    = tid >> 5;       // wave in block: hidden chunk owner
  const int lane   = tid & 31;
  const int half   = lane >> 4;
  const int m      = lane & 15;      // A-row / batch column
  const int batch0 = blockIdx.x * 16;

  // ---- A tiles (W_hh, f16, WMMA A layout) for gate classes I,F,G,O of this hidden chunk.
  // Gate row for class G: g = 32*G + 16*wib + m ; K = 16*half + 2*j (+1).
  v16h A[4];
#pragma unroll
  for (int G = 0; G < 4; ++G) {
#pragma unroll
    for (int j = 0; j < 8; ++j) {
      int g  = 32 * G + 16 * wib + m;
      int kb = (half << 4) + (j << 1);
      A[G][2 * j]     = (_Float16)W_hh[g * HID + kb];
      A[G][2 * j + 1] = (_Float16)W_hh[g * HID + kb + 1];
    }
  }

  // ---- Packed (W_ih, b_ih+b_hh) per D-layout slot: gate = 32*G + 16*wib + 8*half + r.
  h2 wb[32];
#pragma unroll
  for (int G = 0; G < 4; ++G)
#pragma unroll
    for (int r = 0; r < 8; ++r) {
      int g = 32 * G + 16 * wib + 8 * half + r;
      wb[G * 8 + r][0] = (_Float16)W_ih[g];
      wb[G * 8 + r][1] = (_Float16)(b_ih[g] + b_hh[g]);
    }

  // ---- State (D layout): hidden = 16*wib + 8*half + r, batch = m.
  float c[8], h[8];
#pragma unroll
  for (int r = 0; r < 8; ++r) { c[r] = 0.f; h[r] = 0.f; }

  const float* xsrc = x + (unsigned long long)batch0 * TLEN;

#if USE_TDM
  if (wib == 0) tdm_load_tile(xsrc, (unsigned)(unsigned long long)&xls[0][0]);
#endif

#pragma unroll 1
  for (int ch = 0; ch < NCHUNK; ++ch) {
    float* cur = &xls[ch & 1][0];
#if USE_TDM
    if (wib == 0) __builtin_amdgcn_s_wait_tensorcnt(0);    // only cur's load is outstanding
#else
    stage_chunk(cur, xsrc, ch * CT, tid);
#endif
    __syncthreads();                                       // cur visible to both waves
#if USE_TDM
    if (wib == 0 && ch + 1 < NCHUNK)
      tdm_load_tile(xsrc + (ch + 1) * CT, (unsigned)(unsigned long long)&xls[(ch & 1) ^ 1][0]);
#endif

#pragma unroll 1
    for (int tl = 0; tl < CT; ++tl) {
      const int p = tl & 1;

      // 1) Publish our h half as packed f16 (4 dwords/lane), then signal.
      unsigned pk[4];
#pragma unroll
      for (int j = 0; j < 4; ++j) {
        h2 a;
        a[0] = (_Float16)h[2 * j];
        a[1] = (_Float16)h[2 * j + 1];
        pk[j] = __builtin_bit_cast(unsigned, a);
      }
      *(uint4*)&xch[p][wib * 2 + half][m][0] = make_uint4(pk[0], pk[1], pk[2], pk[3]);
      BAR_SIGNAL();

      // 2) Independent work while the barrier settles: C-init = x*W_ih + bias.
      float xv = cur[m * ROWW + tl];
      v8f cI, cF, cG, cO;
#pragma unroll
      for (int r = 0; r < 8; ++r) {
        cI[r] = (float)wb[0 * 8 + r][0] * xv + (float)wb[0 * 8 + r][1];
        cF[r] = (float)wb[1 * 8 + r][0] * xv + (float)wb[1 * 8 + r][1];
        cG[r] = (float)wb[2 * 8 + r][0] * xv + (float)wb[2 * 8 + r][1];
        cO[r] = (float)wb[3 * 8 + r][0] * xv + (float)wb[3 * 8 + r][1];
      }
      BAR_WAIT();

      // 3) Gather the full h^T B operand. B layout: lane half H holds K=16H..16H+15;
      //    K=16H+2j+e from (wave H, lane-half 0), K=16H+8+2j+e from (wave H, lane-half 1).
      uint4 lo = *(const uint4*)&xch[p][half * 2 + 0][m][0];
      uint4 hi = *(const uint4*)&xch[p][half * 2 + 1][m][0];
      v8u hbu = {lo.x, lo.y, lo.z, lo.w, hi.x, hi.y, hi.z, hi.w};
      v16h hbv = __builtin_bit_cast(v16h, hbu);

      // 4) Gates for our hidden chunk: 4 x WMMA 16x16x32 f16.
      v8f dI = __builtin_amdgcn_wmma_f32_16x16x32_f16(false, A[0], false, hbv, (short)0, cI, false, false);
      v8f dF = __builtin_amdgcn_wmma_f32_16x16x32_f16(false, A[1], false, hbv, (short)0, cF, false, false);
      v8f dG = __builtin_amdgcn_wmma_f32_16x16x32_f16(false, A[2], false, hbv, (short)0, cG, false, false);
      v8f dO = __builtin_amdgcn_wmma_f32_16x16x32_f16(false, A[3], false, hbv, (short)0, cO, false, false);

      // 5) Pointwise LSTM cell update for our 8 hidden values per lane.
#pragma unroll
      for (int r = 0; r < 8; ++r) {
        float ig = fast_sigm(dI[r]);
        float fg = fast_sigm(dF[r]);
        float gg = fast_tanh(dG[r]);
        float og = fast_sigm(dO[r]);
        float cn = __builtin_fmaf(fg, c[r], ig * gg);
        c[r] = cn;
        h[r] = og * fast_tanh(cn);
      }
    }
  }

  // ---- Final FC: out[b, o] = sum_h h[b,h] * fc_w[o,h] + fc_b[o]
  float p0 = 0.f, p1 = 0.f;
#pragma unroll
  for (int r = 0; r < 8; ++r) {
    int hid = 16 * wib + 8 * half + r;
    p0 = __builtin_fmaf(h[r], fc_w[hid], p0);
    p1 = __builtin_fmaf(h[r], fc_w[32 + hid], p1);
  }
  fpart[0][tid] = p0;
  fpart[1][tid] = p1;
  __syncthreads();
  if (tid < 16) {
    float o0 = fc_b[0] + fpart[0][tid] + fpart[0][tid + 16] + fpart[0][tid + 32] + fpart[0][tid + 48];
    float o1 = fc_b[1] + fpart[1][tid] + fpart[1][tid + 16] + fpart[1][tid + 32] + fpart[1][tid + 48];
    out[(batch0 + tid) * 2 + 0] = o0;
    out[(batch0 + tid) * 2 + 1] = o1;
  }
}

extern "C" void kernel_launch(void* const* d_in, const int* in_sizes, int n_in,
                              void* d_out, int out_size, void* d_ws, size_t ws_size,
                              hipStream_t stream) {
  const float* x    = (const float*)d_in[0];
  const float* W_ih = (const float*)d_in[1];
  const float* W_hh = (const float*)d_in[2];
  const float* b_ih = (const float*)d_in[3];
  const float* b_hh = (const float*)d_in[4];
  const float* fc_w = (const float*)d_in[5];
  const float* fc_b = (const float*)d_in[6];
  float* out = (float*)d_out;

  dim3 grid(4096 / 16);   // 256 workgroups x 2 waves = 512 waves, 16 batch rows each
  dim3 block(64);
  lstm_wmma_split_kernel<<<grid, block, 0, stream>>>(x, W_ih, W_hh, b_ih, b_hh, fc_w, fc_b, out);
}
